// CrossRQVAE_30322469109854
// MI455X (gfx1250) — matrix operand order repaired and checked by reading, not compiled, not run on hardware
//
#include <hip/hip_runtime.h>
#include <stdint.h>

#define BB 8192
#define ED 256
#define NE 4096
#define NL 3

typedef __attribute__((ext_vector_type(16))) __bf16 v16bf;
typedef __attribute__((ext_vector_type(8)))  float  v8f;

// ---------------- workspace layout (bytes) ----------------
#define WS_RESID 0ULL                      // f32 [BB,ED]        8388608
#define WS_AHI   (8388608ULL)              // bf16 [BB,ED]       4194304
#define WS_ALO   (12582912ULL)             // bf16 [BB,ED]       4194304
#define WS_CBHI  (16777216ULL)             // bf16 [NL,NE,ED]    6291456
#define WS_CBLO  (23068672ULL)             // bf16 [NL,NE,ED]    6291456
#define WS_RR    (29360128ULL)             // f32 [BB]           32768
#define WS_CC    (29392896ULL)             // f32 [NL,NE]        49152
#define WS_AMIN  (29442048ULL)             // u64 [NL,BB]        196608
#define WS_LSUM  (29638656ULL)             // f32 [NL,BB]        98304
// total ~28.4 MB

// ---------------- output layout (float elements) ----------------
// tuple: x_q [B,E], mean_loss, all_idx [B,L], all_oh [B,L,NE], all_logits [B,L,NE]
constexpr size_t O_XQ   = 0;
constexpr size_t O_LOSS = 2097152;
constexpr size_t O_IDX  = 2097153;
constexpr size_t O_OH   = 2121729;
constexpr size_t O_LG   = 102785025;

union Frag {
  v16bf v;
  uint4 u[2];
};

__device__ __forceinline__ unsigned short f2bf(float f) {
  unsigned u = __float_as_uint(f);
  u += 0x7FFFu + ((u >> 16) & 1u);          // round-to-nearest-even
  return (unsigned short)(u >> 16);
}
__device__ __forceinline__ float bf2f(unsigned short h) {
  return __uint_as_float(((unsigned)h) << 16);
}
// order-preserving float -> uint map (total order, works for negatives)
__device__ __forceinline__ unsigned fmap(float f) {
  unsigned u = __float_as_uint(f);
  return (u & 0x80000000u) ? ~u : (u | 0x80000000u);
}
__device__ __forceinline__ v8f wmma_bf16(v16bf a, v16bf b, v8f c) {
  return __builtin_amdgcn_wmma_f32_16x16x32_bf16(false, a, false, b,
                                                 (short)0, c, false, false);
}

// ---------------- prep: codebook -> bf16 hi/lo + row norms ----------------
__global__ void prep_cb(const float* __restrict__ cb,
                        unsigned short* __restrict__ cbhi,
                        unsigned short* __restrict__ cblo,
                        float* __restrict__ cc) {
  const int row = blockIdx.x;               // l*NE + j
  const int t = threadIdx.x;
  const size_t p = (size_t)row * ED + t;
  float v = cb[p];
  unsigned short h = f2bf(v);
  cbhi[p] = h;
  cblo[p] = f2bf(v - bf2f(h));
  __shared__ float s[256];
  s[t] = v * v; __syncthreads();
  for (int st = 128; st > 0; st >>= 1) { if (t < st) s[t] += s[t + st]; __syncthreads(); }
  if (t == 0) cc[row] = s[0];
}

// ---------------- prep: residual = x, bf16 hi/lo + row norms ----------------
__global__ void prep_x(const float* __restrict__ x,
                       float* __restrict__ resid,
                       unsigned short* __restrict__ ahi,
                       unsigned short* __restrict__ alo,
                       float* __restrict__ rr) {
  const int b = blockIdx.x;
  const int t = threadIdx.x;
  const size_t p = (size_t)b * ED + t;
  float v = x[p];
  resid[p] = v;
  unsigned short h = f2bf(v);
  ahi[p] = h;
  alo[p] = f2bf(v - bf2f(h));
  __shared__ float s[256];
  s[t] = v * v; __syncthreads();
  for (int st = 128; st > 0; st >>= 1) { if (t < st) s[t] += s[t + st]; __syncthreads(); }
  if (t == 0) rr[b] = s[0];
}

__global__ void init_k(unsigned long long* __restrict__ amin) {
  int i = blockIdx.x * 256 + threadIdx.x;
  if (i < NL * BB) amin[i] = ~0ULL;
}

// ---------------- fused GEMM-distance + logits + packed argmin ----------------
// C = rr + cc - 2 * residual @ cb^T, split-precision bf16 WMMA (hi*hi + hi*lo + lo*hi)
// block: 256 threads = 8 waves, tile 64(M) x 256(N); wave tile 32 x 64.
__global__ __launch_bounds__(256) void gemm_dist(
    const unsigned short* __restrict__ Ahi, const unsigned short* __restrict__ Alo,
    const unsigned short* __restrict__ Bhi, const unsigned short* __restrict__ Blo,
    const float* __restrict__ rr, const float* __restrict__ ccl,
    float* __restrict__ out, unsigned long long* __restrict__ aminl, int layer)
{
  const int lane  = threadIdx.x & 31;
  const int wid   = threadIdx.x >> 5;
  const int l16   = lane & 15;
  const bool hih  = lane >= 16;
  const int mwave = blockIdx.y * 64 + (wid >> 2) * 32;
  const int nwave = blockIdx.x * 256 + (wid & 3) * 64;
  const int kA = hih ? 8 : 0;     // A frag: lanes 16-31 start at K+8
  const int kB = hih ? 8 : 0;     // B frag: lanes 16-31 start at K+8

  v8f acc[2][4];
  v8f z = {0.f, 0.f, 0.f, 0.f, 0.f, 0.f, 0.f, 0.f};
#pragma unroll
  for (int m = 0; m < 2; ++m)
#pragma unroll
    for (int n = 0; n < 4; ++n) acc[m][n] = z;

  for (int k0 = 0; k0 < ED; k0 += 32) {
    Frag ah[2], al[2];
#pragma unroll
    for (int m = 0; m < 2; ++m) {
      size_t off = (size_t)(mwave + m * 16 + l16) * ED + (k0 + kA);
      ah[m].u[0] = *(const uint4*)(Ahi + off);
      ah[m].u[1] = *(const uint4*)(Ahi + off + 16);
      al[m].u[0] = *(const uint4*)(Alo + off);
      al[m].u[1] = *(const uint4*)(Alo + off + 16);
      if (k0 + 32 < ED) {
        __builtin_prefetch(Ahi + off + 32, 0, 1);   // global_prefetch_b8
        __builtin_prefetch(Alo + off + 32, 0, 1);
      }
    }
#pragma unroll
    for (int n = 0; n < 4; ++n) {
      size_t offb = (size_t)(nwave + n * 16 + l16) * ED + (k0 + kB);
      Frag bh, bl;
      bh.u[0] = *(const uint4*)(Bhi + offb);
      bh.u[1] = *(const uint4*)(Bhi + offb + 16);
      bl.u[0] = *(const uint4*)(Blo + offb);
      bl.u[1] = *(const uint4*)(Blo + offb + 16);
#pragma unroll
      for (int m = 0; m < 2; ++m) {
        acc[m][n] = wmma_bf16(ah[m].v, bh.v, acc[m][n]);  // hi*hi
        acc[m][n] = wmma_bf16(ah[m].v, bl.v, acc[m][n]);  // hi*lo
        acc[m][n] = wmma_bf16(al[m].v, bh.v, acc[m][n]);  // lo*hi
      }
    }
  }

  // epilogue: d = rr + cc - 2*acc; store logits; packed argmin reduce
#pragma unroll
  for (int m = 0; m < 2; ++m) {
    const int rbase = mwave + m * 16 + (hih ? 8 : 0);   // rows rbase..rbase+7
    float rrv[8];
#pragma unroll
    for (int r = 0; r < 8; ++r) rrv[r] = rr[rbase + r];
    unsigned long long key[8];
#pragma unroll
    for (int r = 0; r < 8; ++r) key[r] = ~0ULL;

#pragma unroll
    for (int n = 0; n < 4; ++n) {
      const int j = nwave + n * 16 + l16;
      const float cj = ccl[j];
#pragma unroll
      for (int r = 0; r < 8; ++r) {
        float d = rrv[r] + cj - 2.0f * acc[m][n][r];
        out[O_LG + ((size_t)(rbase + r) * NL + layer) * NE + j] = d;
        unsigned long long kk = ((unsigned long long)fmap(d) << 32) | (unsigned)j;
        if (kk < key[r]) key[r] = kk;
      }
    }
    // reduce across the 16 lanes that share each row, then one atomic per row
#pragma unroll
    for (int r = 0; r < 8; ++r) {
      unsigned long long k = key[r];
#pragma unroll
      for (int msk = 1; msk < 16; msk <<= 1) {
        unsigned long long o = __shfl_xor(k, msk, 16);
        if (o < k) k = o;
      }
      if (l16 == 0) atomicMin(aminl + rbase + r, k);
    }
  }
}

// ---------------- one-hot + gather + residual/loss update (block = one row) ----------------
__global__ void update_k(const float* __restrict__ cbl,
                         float* __restrict__ resid,
                         unsigned short* __restrict__ ahi,
                         unsigned short* __restrict__ alo,
                         float* __restrict__ rr,
                         const unsigned long long* __restrict__ aminl,
                         float* __restrict__ out,
                         float* __restrict__ lsuml, int layer)
{
  const int b = blockIdx.x;
  const int t = threadIdx.x;
  const unsigned idx = (unsigned)(aminl[b] & 0xFFFFFFFFULL);

  // one-hot row (scalar dword stores; region base is odd so no vector stores)
  const size_t ohb = O_OH + ((size_t)b * NL + layer) * NE;
#pragma unroll
  for (int j = t; j < NE; j += 256)
    out[ohb + j] = (j == (int)idx) ? 1.0f : 0.0f;

  const size_t p = (size_t)b * ED + t;
  const float q = cbl[(size_t)idx * ED + t];
  const float r_old = resid[p];
  const float diff = q - r_old;          // (x_q - residual), forward value of both losses
  const float r_new = r_old - q;         // next residual (STE forward = x_q)
  resid[p] = r_new;
  if (layer == 0) out[O_XQ + p] = q; else out[O_XQ + p] += q;

  unsigned short h = f2bf(r_new);
  ahi[p] = h;
  alo[p] = f2bf(r_new - bf2f(h));

  __shared__ float s[256];
  s[t] = diff * diff; __syncthreads();
  for (int st = 128; st > 0; st >>= 1) { if (t < st) s[t] += s[t + st]; __syncthreads(); }
  if (t == 0) lsuml[b] = s[0];
  __syncthreads();
  s[t] = r_new * r_new; __syncthreads();
  for (int st = 128; st > 0; st >>= 1) { if (t < st) s[t] += s[t + st]; __syncthreads(); }
  if (t == 0) {
    rr[b] = s[0];
    out[O_IDX + (size_t)b * NL + layer] = (float)idx;
  }
}

// ---------------- deterministic loss finalize ----------------
__global__ void fin_k(const float* __restrict__ lsum, float* __restrict__ out) {
  const int t = threadIdx.x;
  float a = 0.0f;
  for (int i = t; i < NL * BB; i += 256) a += lsum[i];
  __shared__ float s[256];
  s[t] = a; __syncthreads();
  for (int st = 128; st > 0; st >>= 1) { if (t < st) s[t] += s[t + st]; __syncthreads(); }
  // loss_l = (1+BETA) * SSE_l / (B*E); mean over layers
  if (t == 0) out[O_LOSS] = s[0] * (1.25f / ((float)BB * (float)ED * (float)NL));
}

extern "C" void kernel_launch(void* const* d_in, const int* in_sizes, int n_in,
                              void* d_out, int out_size, void* d_ws, size_t ws_size,
                              hipStream_t stream)
{
  (void)in_sizes; (void)n_in; (void)out_size; (void)ws_size;
  const float* x  = (const float*)d_in[0];
  const float* cb = (const float*)d_in[1];
  float* out = (float*)d_out;
  char* ws = (char*)d_ws;

  float*              resid = (float*)(ws + WS_RESID);
  unsigned short*     ahi   = (unsigned short*)(ws + WS_AHI);
  unsigned short*     alo   = (unsigned short*)(ws + WS_ALO);
  unsigned short*     cbhi  = (unsigned short*)(ws + WS_CBHI);
  unsigned short*     cblo  = (unsigned short*)(ws + WS_CBLO);
  float*              rr    = (float*)(ws + WS_RR);
  float*              cc    = (float*)(ws + WS_CC);
  unsigned long long* amin  = (unsigned long long*)(ws + WS_AMIN);
  float*              lsum  = (float*)(ws + WS_LSUM);

  prep_cb<<<NL * NE, 256, 0, stream>>>(cb, cbhi, cblo, cc);
  prep_x <<<BB,      256, 0, stream>>>(x, resid, ahi, alo, rr);
  init_k <<<(NL * BB + 255) / 256, 256, 0, stream>>>(amin);

  for (int l = 0; l < NL; ++l) {
    gemm_dist<<<dim3(16, 128), 256, 0, stream>>>(
        ahi, alo,
        cbhi + (size_t)l * NE * ED, cblo + (size_t)l * NE * ED,
        rr, cc + (size_t)l * NE,
        out, amin + (size_t)l * BB, l);
    update_k<<<BB, 256, 0, stream>>>(
        cb + (size_t)l * NE * ED, resid, ahi, alo, rr,
        amin + (size_t)l * BB, out, lsum + (size_t)l * BB, l);
  }
  fin_k<<<1, 256, 0, stream>>>(lsum, out);
}
